// SelfInteraction_38671885533629
// MI455X (gfx1250) — compile-verified, matrix operand order checked by reading, not run on hardware
//
#include <hip/hip_runtime.h>
#include <math.h>

// ---------------------------------------------------------------------------
// SelfInteraction (2 layers) for MI455X / gfx1250, wave32 + WMMA f16->f32.
// RT=32 rows/block, dual-row-tile B reuse (each B fragment -> 2 WMMAs).
// B addressing: per-column-tile base pointer + constant K-step strides so the
// packed-B loads use immediate offsets (no per-step 64-bit address VALU, no
// WMMA->VALU WAR hazard NOPs in the inner loop).
//
// Workspace layout (bytes), total ~73.2 MB:
//   [0        , 1900544 )  packed f16 weights (2 layers)
//   [1900544  , 14483456)  scg  : N x 384 f16 (gated invariants)
//   [14483456 , 39649280)  vecg : 3 x N x 256 f16 (gated equivariants, d-major)
//   [39649280 , 48037888)  ds   : N x 128 f32
//   [48037888 , 73203712)  dv   : N x 384 f32
// ---------------------------------------------------------------------------

#define NROWS 16384
#define RT    32     // rows per block tile (2 x 16-row WMMA tiles)

typedef float    v8f    __attribute__((ext_vector_type(8)));
typedef _Float16 half8  __attribute__((ext_vector_type(8)));
typedef _Float16 half16 __attribute__((ext_vector_type(16)));

#define RS384 0.051031036307982884f   // 1/sqrt(384)
#define RS256 0.0625f                 // 1/sqrt(256)
#define ISQ3  0.57735026918962576f    // 1/sqrt(3)
#define SQ2   1.4142135623730951f

// per-layer packed-weight half offsets
#define PW1 0
#define PW2 147456
#define PW0 393216
#define PWV 442368
#define PLW 475136   // halves per layer

__device__ inline half16 combine16(half8 lo, half8 hi) {
  half16 r;
#pragma unroll
  for (int i = 0; i < 8; ++i) { r[i] = lo[i]; r[i + 8] = hi[i]; }
  return r;
}

// ---------------------------------------------------------------------------
// Weight pack: fp32 (K x N row-major, row stride srcStride) -> f16 WMMA B-block
// layout: block(kt,nt) = 512 halves; lane L owns halves [L*16..L*16+15]:
//   n = 16*nt + (L&15);  k = 32*kt + 8*(L>>4) + (h&7) + ((h>>3)<<4)
// ---------------------------------------------------------------------------
__global__ void kw_pack(const float* __restrict__ src, _Float16* __restrict__ dst,
                        int Ktot, int Npack, int srcStride) {
  int idx = blockIdx.x * 256 + threadIdx.x;
  int total = Ktot * Npack;
  if (idx >= total) return;
  int blk    = idx >> 9;
  int within = idx & 511;
  int lane   = within >> 4;
  int h      = within & 15;
  int nb     = Npack >> 4;
  int kt     = blk / nb;
  int nt     = blk - kt * nb;
  int k = 32 * kt + 8 * (lane >> 4) + (h & 7) + ((h >> 3) << 4);
  int n = 16 * nt + (lane & 15);
  dst[idx] = (_Float16)src[(size_t)k * srcStride + n];
}

// ---------------------------------------------------------------------------
// Kernel A: tensor-square features + GEMM1(silu) + GEMM2 + gating.
// ---------------------------------------------------------------------------
__global__ void __launch_bounds__(256)
ka_fused(const float* __restrict__ state,
         const _Float16* __restrict__ w1p,
         const _Float16* __restrict__ w2p,
         _Float16* __restrict__ scg,
         _Float16* __restrict__ vecg) {
  __shared__ _Float16 sSC[RT][392];   // invariants (then gated in-place)
  __shared__ _Float16 sH [RT][392];   // hidden after silu
  __shared__ float    sG [RT][256];   // vec gates

  const int tid  = threadIdx.x;
  const int wave = tid >> 5;
  const int lane = tid & 31;
  const int l15  = lane & 15;
  const int kh   = lane >> 4;
  const int rowbase = blockIdx.x * RT;

  // stage sc = [s, s*s, |v|^2/sqrt3] as f16
  for (int i = tid; i < RT * 384; i += 256) {
    int m = i / 384, c = i - m * 384;
    const float* xr = state + (size_t)(rowbase + m) * 512;
    float val;
    if (c < 128) {
      val = xr[c];
    } else if (c < 256) {
      float s = xr[c - 128];
      val = s * s;
    } else {
      const float* vp = xr + 128 + (c - 256) * 3;
      val = (vp[0] * vp[0] + vp[1] * vp[1] + vp[2] * vp[2]) * ISQ3;
    }
    sSC[m][c] = (_Float16)val;
  }
  __syncthreads();

  // GEMM1: H = silu(SC @ W1 * rs384), 32x384 @ 384x384, dual row tiles per B
  for (int ct = wave; ct < 24; ct += 8) {
    if (ct + 8 < 24)   // prefetch next column tile's first B block
      __builtin_prefetch((const void*)(w1p + (size_t)(ct + 8) * 512 +
                                       (size_t)lane * 16), 0, 0);
    const half16* bp = (const half16*)w1p + (size_t)ct * 32 + lane;
    v8f acc0 = {}, acc1 = {};
#pragma unroll
    for (int ks = 0; ks < 12; ++ks) {
      half16 B = bp[ks * (24 * 32)];   // +24576 B per K step (imm offset)
      half16 A0 = combine16(*(const half8*)&sSC[l15][32 * ks + 8 * kh],
                            *(const half8*)&sSC[l15][32 * ks + 8 * kh + 16]);
      half16 A1 = combine16(*(const half8*)&sSC[16 + l15][32 * ks + 8 * kh],
                            *(const half8*)&sSC[16 + l15][32 * ks + 8 * kh + 16]);
      acc0 = __builtin_amdgcn_wmma_f32_16x16x32_f16(false, A0, false, B,
                                                    (short)0, acc0, false, false);
      acc1 = __builtin_amdgcn_wmma_f32_16x16x32_f16(false, A1, false, B,
                                                    (short)0, acc1, false, false);
    }
#pragma unroll
    for (int r = 0; r < 8; ++r) {
      float x0 = acc0[r] * RS384;
      float x1 = acc1[r] * RS384;
      sH[r + 8 * kh][ct * 16 + l15]      = (_Float16)(x0 / (1.0f + __expf(-x0)));
      sH[16 + r + 8 * kh][ct * 16 + l15] = (_Float16)(x1 / (1.0f + __expf(-x1)));
    }
  }
  __syncthreads();

  // GEMM2: G = H @ W2[:, :640] * rs384 ; gate sc in place, stash vec gates
  for (int ct = wave; ct < 40; ct += 8) {
    if (ct + 8 < 40)
      __builtin_prefetch((const void*)(w2p + (size_t)(ct + 8) * 512 +
                                       (size_t)lane * 16), 0, 0);
    const half16* bp = (const half16*)w2p + (size_t)ct * 32 + lane;
    v8f acc0 = {}, acc1 = {};
#pragma unroll
    for (int ks = 0; ks < 12; ++ks) {
      half16 B = bp[ks * (40 * 32)];   // +40960 B per K step (imm offset)
      half16 A0 = combine16(*(const half8*)&sH[l15][32 * ks + 8 * kh],
                            *(const half8*)&sH[l15][32 * ks + 8 * kh + 16]);
      half16 A1 = combine16(*(const half8*)&sH[16 + l15][32 * ks + 8 * kh],
                            *(const half8*)&sH[16 + l15][32 * ks + 8 * kh + 16]);
      acc0 = __builtin_amdgcn_wmma_f32_16x16x32_f16(false, A0, false, B,
                                                    (short)0, acc0, false, false);
      acc1 = __builtin_amdgcn_wmma_f32_16x16x32_f16(false, A1, false, B,
                                                    (short)0, acc1, false, false);
    }
#pragma unroll
    for (int r = 0; r < 8; ++r) {
      int col = ct * 16 + l15;
      float g0v = acc0[r] * RS384;
      float g1v = acc1[r] * RS384;
      int m0 = r + 8 * kh;
      int m1 = 16 + r + 8 * kh;
      if (col < 384) {
        sSC[m0][col] = (_Float16)((float)sSC[m0][col] * g0v);
        sSC[m1][col] = (_Float16)((float)sSC[m1][col] * g1v);
      } else {
        sG[m0][col - 384] = g0v;
        sG[m1][col - 384] = g1v;
      }
    }
  }
  __syncthreads();

  // flush gated invariants (coalesced f16)
  for (int i = tid; i < RT * 384; i += 256) {
    int m = i / 384, c = i - m * 384;
    scg[(size_t)(rowbase + m) * 384 + c] = sSC[m][c];
  }
  // flush gated equivariants: vec = [v, sqrt2*s*v] * g_v  (d-major matrices)
  for (int i = tid; i < 3 * RT * 256; i += 256) {
    int d   = i / (RT * 256);
    int rem = i - d * RT * 256;
    int m   = rem >> 8;
    int c   = rem & 255;
    const float* xr = state + (size_t)(rowbase + m) * 512;
    float g  = sG[m][c];
    int  ch  = (c < 128) ? c : (c - 128);
    float vv = xr[128 + ch * 3 + d];
    float val = (c < 128) ? vv * g : SQ2 * xr[ch] * vv * g;
    vecg[(size_t)d * ((size_t)NROWS * 256) + (size_t)(rowbase + m) * 256 + c] =
        (_Float16)val;
  }
}

// ---------------------------------------------------------------------------
// Kernel B: out_s = SCg @ W0 * rs384 ; out_v[d] = VECg[d] @ WV * rs256
// ---------------------------------------------------------------------------
__global__ void __launch_bounds__(256)
kb_out(const _Float16* __restrict__ scg,
       const _Float16* __restrict__ vecg,
       const _Float16* __restrict__ w0p,
       const _Float16* __restrict__ wvp,
       float* __restrict__ dsb,
       float* __restrict__ dvb) {
  __shared__ _Float16 sA[RT][392];       // gated invariants
  __shared__ _Float16 sV[3][RT][264];    // gated equivariants

  const int tid  = threadIdx.x;
  const int wave = tid >> 5;
  const int lane = tid & 31;
  const int l15  = lane & 15;
  const int kh   = lane >> 4;
  const int rowbase = blockIdx.x * RT;

  for (int i = tid; i < RT * 384; i += 256) {
    int m = i / 384, c = i - m * 384;
    sA[m][c] = scg[(size_t)(rowbase + m) * 384 + c];
  }
  for (int i = tid; i < 3 * RT * 256; i += 256) {
    int d   = i / (RT * 256);
    int rem = i - d * RT * 256;
    int m   = rem >> 8;
    int c   = rem & 255;
    sV[d][m][c] =
        vecg[(size_t)d * ((size_t)NROWS * 256) + (size_t)(rowbase + m) * 256 + c];
  }
  __syncthreads();

  for (int t = wave; t < 32; t += 8) {
    v8f acc0 = {}, acc1 = {};
    if (t < 8) {                       // out_s column tile, K=384
      int ct = t;
      const half16* bp = (const half16*)w0p + (size_t)ct * 32 + lane;
#pragma unroll
      for (int ks = 0; ks < 12; ++ks) {
        half16 B = bp[ks * (8 * 32)];  // +8192 B per K step (imm offset)
        half16 A0 = combine16(*(const half8*)&sA[l15][32 * ks + 8 * kh],
                              *(const half8*)&sA[l15][32 * ks + 8 * kh + 16]);
        half16 A1 = combine16(*(const half8*)&sA[16 + l15][32 * ks + 8 * kh],
                              *(const half8*)&sA[16 + l15][32 * ks + 8 * kh + 16]);
        acc0 = __builtin_amdgcn_wmma_f32_16x16x32_f16(false, A0, false, B,
                                                      (short)0, acc0, false, false);
        acc1 = __builtin_amdgcn_wmma_f32_16x16x32_f16(false, A1, false, B,
                                                      (short)0, acc1, false, false);
      }
#pragma unroll
      for (int r = 0; r < 8; ++r) {
        int m = r + 8 * kh;
        dsb[(size_t)(rowbase + m) * 128 + ct * 16 + l15]      = acc0[r] * RS384;
        dsb[(size_t)(rowbase + 16 + m) * 128 + ct * 16 + l15] = acc1[r] * RS384;
      }
    } else {                           // out_v column tile, K=256
      int u = t - 8, d = u >> 3, ct = u & 7;
      const half16* bp = (const half16*)wvp + (size_t)ct * 32 + lane;
#pragma unroll
      for (int ks = 0; ks < 8; ++ks) {
        half16 B = bp[ks * (8 * 32)];  // +8192 B per K step (imm offset)
        half16 A0 = combine16(*(const half8*)&sV[d][l15][32 * ks + 8 * kh],
                              *(const half8*)&sV[d][l15][32 * ks + 8 * kh + 16]);
        half16 A1 = combine16(*(const half8*)&sV[d][16 + l15][32 * ks + 8 * kh],
                              *(const half8*)&sV[d][16 + l15][32 * ks + 8 * kh + 16]);
        acc0 = __builtin_amdgcn_wmma_f32_16x16x32_f16(false, A0, false, B,
                                                      (short)0, acc0, false, false);
        acc1 = __builtin_amdgcn_wmma_f32_16x16x32_f16(false, A1, false, B,
                                                      (short)0, acc1, false, false);
      }
#pragma unroll
      for (int r = 0; r < 8; ++r) {
        int m = r + 8 * kh;
        int c = ct * 16 + l15;
        dvb[(size_t)(rowbase + m) * 384 + c * 3 + d]      = acc0[r] * RS256;
        dvb[(size_t)(rowbase + 16 + m) * 384 + c * 3 + d] = acc1[r] * RS256;
      }
    }
  }
}

// ---------------------------------------------------------------------------
// Kernel C: residual add + (optional) equivariant layernorm. One wave per row.
// ---------------------------------------------------------------------------
__global__ void __launch_bounds__(256)
kc_resln(const float* statein,
         const float* __restrict__ dsb,
         const float* __restrict__ dvb,
         const float* __restrict__ g0,
         const float* __restrict__ g1,
         float* out, int apply_ln) {
  const int wave = threadIdx.x >> 5;
  const int lane = threadIdx.x & 31;
  const size_t row = (size_t)blockIdx.x * 8 + wave;

  const float* xr = statein + row * 512;
  float sv[4], vv[12];
  float ssum = 0.f, ssq = 0.f, vsq = 0.f;
#pragma unroll
  for (int j = 0; j < 4; ++j) {
    int c = lane + 32 * j;
    sv[j] = xr[c] + dsb[row * 128 + c];
    ssum += sv[j];
    ssq  += sv[j] * sv[j];
  }
#pragma unroll
  for (int j = 0; j < 12; ++j) {
    int idx = lane + 32 * j;
    vv[j] = xr[128 + idx] + dvb[row * 384 + idx];
    vsq  += vv[j] * vv[j];
  }
#pragma unroll
  for (int m = 16; m >= 1; m >>= 1) {   // wave32 butterfly reduction
    ssum += __shfl_xor(ssum, m, 32);
    ssq  += __shfl_xor(ssq,  m, 32);
    vsq  += __shfl_xor(vsq,  m, 32);
  }
  float* orow = out + row * 512;
  if (apply_ln) {
    float mu   = ssum * (1.0f / 128.0f);
    float sdin = 1.0f / sqrtf(ssq * (1.0f / 128.0f) - mu * mu + 1e-6f);
    float rinv = rsqrtf(vsq * (1.0f / 384.0f) + 1e-6f);
#pragma unroll
    for (int j = 0; j < 4; ++j) {
      int c = lane + 32 * j;
      orow[c] = (sv[j] - mu) * sdin * g0[c];
    }
#pragma unroll
    for (int j = 0; j < 12; ++j) {
      int idx = lane + 32 * j;
      orow[128 + idx] = vv[j] * rinv * g1[idx / 3];
    }
  } else {
#pragma unroll
    for (int j = 0; j < 4; ++j) orow[lane + 32 * j] = sv[j];
#pragma unroll
    for (int j = 0; j < 12; ++j) orow[128 + lane + 32 * j] = vv[j];
  }
}

// ---------------------------------------------------------------------------
extern "C" void kernel_launch(void* const* d_in, const int* in_sizes, int n_in,
                              void* d_out, int out_size, void* d_ws, size_t ws_size,
                              hipStream_t stream) {
  const float* x  = (const float*)d_in[0];
  const float* w1 = (const float*)d_in[1];   // (2,384,384)
  const float* w2 = (const float*)d_in[2];   // (2,384,768) -> use cols [0,640)
  const float* w0 = (const float*)d_in[3];   // (2,384,128)
  const float* wv = (const float*)d_in[4];   // (2,256,128)
  const float* g0 = (const float*)d_in[5];
  const float* g1 = (const float*)d_in[6];
  float* out = (float*)d_out;

  _Float16* wpack = (_Float16*)d_ws;
  _Float16* scg   = (_Float16*)((char*)d_ws + 1900544);
  _Float16* vecg  = (_Float16*)((char*)d_ws + 14483456);
  float*    dsb   = (float*)((char*)d_ws + 39649280);
  float*    dvb   = (float*)((char*)d_ws + 48037888);

  // pack all weights (both layers) fp32 -> f16 WMMA-B block layout
  for (int L = 0; L < 2; ++L) {
    _Float16* base = wpack + (size_t)L * PLW;
    kw_pack<<<(384 * 384 + 255) / 256, 256, 0, stream>>>(
        w1 + (size_t)L * 384 * 384, base + PW1, 384, 384, 384);
    kw_pack<<<(384 * 640 + 255) / 256, 256, 0, stream>>>(
        w2 + (size_t)L * 384 * 768, base + PW2, 384, 640, 768);
    kw_pack<<<(384 * 128 + 255) / 256, 256, 0, stream>>>(
        w0 + (size_t)L * 384 * 128, base + PW0, 384, 128, 128);
    kw_pack<<<(256 * 128 + 255) / 256, 256, 0, stream>>>(
        wv + (size_t)L * 256 * 128, base + PWV, 256, 128, 128);
  }

  const int nblk = NROWS / RT;   // 512
  for (int L = 0; L < 2; ++L) {
    const float* state = (L == 0) ? x : out;
    _Float16* base = wpack + (size_t)L * PLW;
    ka_fused<<<nblk, 256, 0, stream>>>(state, base + PW1, base + PW2, scg, vecg);
    kb_out<<<nblk, 256, 0, stream>>>(scg, vecg, base + PW0, base + PWV, dsb, dvb);
    kc_resln<<<NROWS / 8, 256, 0, stream>>>(state, dsb, dvb, g0, g1, out,
                                            (L == 0) ? 1 : 0);
  }
}
